// LinTrans_29540785062082
// MI455X (gfx1250) — matrix operand order repaired and checked by reading, not compiled, not run on hardware
//
#include <hip/hip_runtime.h>
#include <hip/hip_bf16.h>

// ---------------------------------------------------------------------------
// Shapes
// ---------------------------------------------------------------------------
#define N_ROWS 32768
#define D_IN   3000
#define H0     1024
#define H1     256
#define NCLUST 20
#define KP1    3008   // D_IN padded up to BK multiple (zero-filled in pre-pass)
#define KP2    1024

// GEMM tiling: 128x128 tile, BK=32, 256 threads = 8 wave32 (2x4), each wave
// owns 64x32 = 4x2 WMMA accumulators. Double-buffered LDS.
#define BM 128
#define BN 128
#define BK 32
#define PITCH 40   // LDS row pitch in bf16 (BK + 8 pad, keeps b128 alignment)

typedef __attribute__((ext_vector_type(16))) __bf16 v16bf;
typedef __attribute__((ext_vector_type(8)))  __bf16 v8bf;
typedef __attribute__((ext_vector_type(8)))  float  v8f;
typedef __attribute__((ext_vector_type(4)))  float  v4f;

union U16bf { v16bf v; v8bf h[2]; };

__device__ __forceinline__ v16bf frag_from(const __bf16* p0, const __bf16* p1) {
  U16bf u;
  u.h[0] = *(const v8bf*)p0;   // ds_load_b128
  u.h[1] = *(const v8bf*)p1;   // ds_load_b128
  return u.v;
}

// gfx1250 async global->LDS copy (GVS mode), tracked with ASYNCcnt.
__device__ __forceinline__ unsigned lds_off(const void* p) {
  // generic LDS address: bits[31:0] are the wave-relative LDS byte offset
  return (unsigned)(unsigned long long)(uintptr_t)p;
}
__device__ __forceinline__ void async_b128(unsigned lds_addr,
                                           unsigned long long base, unsigned voff) {
  asm volatile("global_load_async_to_lds_b128 %0, %1, %2"
               :: "v"(lds_addr), "v"(voff), "s"(base) : "memory");
}
__device__ __forceinline__ void async_wait_all() {
  asm volatile("s_wait_asynccnt 0x0" ::: "memory");
}

// ---------------------------------------------------------------------------
// Pre-pass: transpose + fp32->bf16 (+ zero pad K..KP) :  out[n][k] = in[k][n]
// ---------------------------------------------------------------------------
__global__ __launch_bounds__(256)
void transpose_cvt_kernel(const float* __restrict__ in, __bf16* __restrict__ out,
                          int K, int N, int KP)
{
  __shared__ float t[32][33];
  const int k0 = blockIdx.x * 32;
  const int n0 = blockIdx.y * 32;
  const int tx = threadIdx.x & 31;
  const int ty = threadIdx.x >> 5;   // 0..7
  #pragma unroll
  for (int r = 0; r < 32; r += 8) {
    const int k = k0 + ty + r;
    t[ty + r][tx] = (k < K) ? in[(size_t)k * N + n0 + tx] : 0.0f;
  }
  __syncthreads();
  #pragma unroll
  for (int r = 0; r < 32; r += 8) {
    const int n = n0 + ty + r;
    out[(size_t)n * KP + k0 + tx] = (__bf16)t[tx][ty + r];
  }
}

// ---------------------------------------------------------------------------
// Kernel 1: h = silu(x @ W1 + b1).  x fp32 [N,3000], W1T bf16 [1024,3008],
//           h bf16 [N,1024].  blockIdx.x = column block (fastest) for L2 reuse.
// ---------------------------------------------------------------------------
__global__ __launch_bounds__(256)
void gemm1_silu_kernel(const float* __restrict__ x, const __bf16* __restrict__ W1T,
                       const float* __restrict__ b1, __bf16* __restrict__ hbuf)
{
  __shared__ __align__(16) __bf16 sA[2][BM * PITCH];   // [m][k]
  __shared__ __align__(16) __bf16 sB[2][BN * PITCH];   // [n][k] (pre-transposed)

  const int tid  = threadIdx.x;
  const int lane = tid & 31;
  const int wave = tid >> 5;
  const int mbase = (wave & 1) * 64;
  const int nbase = (wave >> 1) * 32;

  const int colBlk = blockIdx.x * BN;   // fastest dim = columns
  const int rowBlk = blockIdx.y * BM;

  v8f acc[4][2] = {};

  const int ar = tid >> 1;            // A tile row 0..127
  const int ac = (tid & 1) * 16;      // 0/16 k-group
  const int br = tid >> 1;            // B tile n-row 0..127
  const int bc = (tid & 1) * 16;      // 0/16 k-group

  const unsigned long long w1base = (unsigned long long)(uintptr_t)W1T;
  const unsigned bvoff = ((unsigned)(colBlk + br) * KP1 + (unsigned)bc) * 2u;

  auto stageB = [&](int ks, int buf) {   // async bf16 b128 copies
    unsigned lds = lds_off(&sB[buf][br * PITCH + bc]);
    unsigned long long base = w1base + (unsigned long long)(ks * BK) * 2ull;
    async_b128(lds,      base, bvoff);
    async_b128(lds + 16, base, bvoff + 16u);
  };
  auto stageA = [&](int ks, int buf) {   // fp32 load + cvt + ds_store
    const int k0 = ks * BK;
    const float* src = x + (size_t)(rowBlk + ar) * D_IN + k0 + ac;
    float v[16];
    if (k0 + ac + 16 <= D_IN) {
      const v4f* s4 = (const v4f*)src;
      #pragma unroll
      for (int t = 0; t < 4; ++t) {
        v4f tv = s4[t];
        v[4*t+0] = tv[0]; v[4*t+1] = tv[1]; v[4*t+2] = tv[2]; v[4*t+3] = tv[3];
      }
    } else {
      #pragma unroll
      for (int t = 0; t < 16; ++t)
        v[t] = (k0 + ac + t < D_IN) ? src[t] : 0.0f;
    }
    v8bf p0, p1;
    #pragma unroll
    for (int t = 0; t < 8; ++t) { p0[t] = (__bf16)v[t]; p1[t] = (__bf16)v[8 + t]; }
    *(v8bf*)&sA[buf][ar * PITCH + ac]     = p0;
    *(v8bf*)&sA[buf][ar * PITCH + ac + 8] = p1;
  };

  const int nsteps = KP1 / BK;   // 94
  stageB(0, 0);
  stageA(0, 0);

  for (int ks = 0; ks < nsteps; ++ks) {
    const int cur = ks & 1;
    async_wait_all();        // my async copies for buf `cur` have landed
    __syncthreads();         // everyone's staging for buf `cur` visible

    if (ks + 1 < nsteps) {   // overlap next tile's traffic with this tile's WMMAs
      stageB(ks + 1, cur ^ 1);
      stageA(ks + 1, cur ^ 1);
    }

    const int mrow = lane & 15;
    const int kb8  = (lane >> 4) << 3;   // A chunks: kb8, kb8+16
    const int kb16 = (lane >> 4) << 4;   // B chunks: kb8, kb8+16 (same as A)
    v16bf afr[4], bfr[2];
    #pragma unroll
    for (int i = 0; i < 4; ++i) {
      const __bf16* base = &sA[cur][(mbase + 16*i + mrow) * PITCH + kb8];
      afr[i] = frag_from(base, base + 16);
    }
    #pragma unroll
    for (int j = 0; j < 2; ++j) {
      const __bf16* base = &sB[cur][(nbase + 16*j + mrow) * PITCH + kb8];
      bfr[j] = frag_from(base, base + 16);
    }
    #pragma unroll
    for (int i = 0; i < 4; ++i)
      #pragma unroll
      for (int j = 0; j < 2; ++j)
        acc[i][j] = __builtin_amdgcn_wmma_f32_16x16x32_bf16(
            false, afr[i], false, bfr[j], (short)0, acc[i][j], false, false);
  }

  // epilogue: bias + SiLU -> bf16
  const int mrow = lane & 15;
  #pragma unroll
  for (int j = 0; j < 2; ++j) {
    const int gcol = colBlk + nbase + 16*j + mrow;
    const float bias = b1[gcol];
    #pragma unroll
    for (int i = 0; i < 4; ++i) {
      const int rbase = rowBlk + mbase + 16*i + ((lane >> 4) << 3);
      #pragma unroll
      for (int e = 0; e < 8; ++e) {
        float s = acc[i][j][e] + bias;
        float a = s / (1.0f + __expf(-s));
        hbuf[(size_t)(rbase + e) * H0 + gcol] = (__bf16)a;
      }
    }
  }
}

// ---------------------------------------------------------------------------
// Kernel 2: z = h @ W2 + b2.  h bf16 [N,1024], W2T bf16 [256,1024], z fp32.
// Fully async staging (both operands bf16), double buffered.
// ---------------------------------------------------------------------------
__global__ __launch_bounds__(256)
void gemm2_kernel(const __bf16* __restrict__ hbuf, const __bf16* __restrict__ W2T,
                  const float* __restrict__ b2, float* __restrict__ z)
{
  __shared__ __align__(16) __bf16 sA[2][BM * PITCH];
  __shared__ __align__(16) __bf16 sB[2][BN * PITCH];

  const int tid  = threadIdx.x;
  const int lane = tid & 31;
  const int wave = tid >> 5;
  const int mbase = (wave & 1) * 64;
  const int nbase = (wave >> 1) * 32;

  const int colBlk = blockIdx.x * BN;   // 2 col blocks share h row tile via L2
  const int rowBlk = blockIdx.y * BM;

  v8f acc[4][2] = {};

  const int ar = tid >> 1;
  const int ac = (tid & 1) * 16;
  const int br = tid >> 1;
  const int bc = (tid & 1) * 16;

  const unsigned long long habase = (unsigned long long)(uintptr_t)hbuf;
  const unsigned long long w2base = (unsigned long long)(uintptr_t)W2T;
  const unsigned avoff = ((unsigned)(rowBlk + ar) * H0  + (unsigned)ac) * 2u;
  const unsigned bvoff = ((unsigned)(colBlk + br) * KP2 + (unsigned)bc) * 2u;

  auto stage = [&](int ks, int buf) {
    const unsigned long long koff = (unsigned long long)(ks * BK) * 2ull;
    unsigned la = lds_off(&sA[buf][ar * PITCH + ac]);
    unsigned lb = lds_off(&sB[buf][br * PITCH + bc]);
    async_b128(la,      habase + koff, avoff);
    async_b128(la + 16, habase + koff, avoff + 16u);
    async_b128(lb,      w2base + koff, bvoff);
    async_b128(lb + 16, w2base + koff, bvoff + 16u);
  };

  const int nsteps = H0 / BK;   // 32
  stage(0, 0);

  for (int ks = 0; ks < nsteps; ++ks) {
    const int cur = ks & 1;
    async_wait_all();
    __syncthreads();
    if (ks + 1 < nsteps) stage(ks + 1, cur ^ 1);

    const int mrow = lane & 15;
    const int kb8  = (lane >> 4) << 3;
    const int kb16 = (lane >> 4) << 4;
    v16bf afr[4], bfr[2];
    #pragma unroll
    for (int i = 0; i < 4; ++i) {
      const __bf16* base = &sA[cur][(mbase + 16*i + mrow) * PITCH + kb8];
      afr[i] = frag_from(base, base + 16);
    }
    #pragma unroll
    for (int j = 0; j < 2; ++j) {
      const __bf16* base = &sB[cur][(nbase + 16*j + mrow) * PITCH + kb8];
      bfr[j] = frag_from(base, base + 16);
    }
    #pragma unroll
    for (int i = 0; i < 4; ++i)
      #pragma unroll
      for (int j = 0; j < 2; ++j)
        acc[i][j] = __builtin_amdgcn_wmma_f32_16x16x32_bf16(
            false, afr[i], false, bfr[j], (short)0, acc[i][j], false, false);
  }

  const int mrow = lane & 15;
  #pragma unroll
  for (int j = 0; j < 2; ++j) {
    const int gcol = colBlk + nbase + 16*j + mrow;
    const float bias = b2[gcol];
    #pragma unroll
    for (int i = 0; i < 4; ++i) {
      const int rbase = rowBlk + mbase + 16*i + ((lane >> 4) << 3);
      #pragma unroll
      for (int e = 0; e < 8; ++e)
        z[(size_t)(rbase + e) * H1 + gcol] = acc[i][j][e] + bias;
    }
  }
}

// ---------------------------------------------------------------------------
// Kernel 3: Student-t soft assignment (V=1 -> q = 1/(1+d2), row-normalized)
// ---------------------------------------------------------------------------
__global__ __launch_bounds__(256)
void soft_assign_kernel(const float* __restrict__ z, const float* __restrict__ clusters,
                        float* __restrict__ q)
{
  __shared__ float cl[NCLUST * H1];
  for (int i = threadIdx.x; i < NCLUST * H1; i += 256)
    cl[i] = clusters[i];
  __syncthreads();

  const int lane = threadIdx.x & 31;
  const int wave = threadIdx.x >> 5;
  const int row  = blockIdx.x * 8 + wave;

  const float* zr = z + (size_t)row * H1;
  float zv[8];
  #pragma unroll
  for (int j = 0; j < 8; ++j) zv[j] = zr[lane + 32 * j];

  float qmine = 0.0f, qsum = 0.0f;
  #pragma unroll
  for (int k = 0; k < NCLUST; ++k) {
    float d = 0.0f;
    #pragma unroll
    for (int j = 0; j < 8; ++j) {
      float t = zv[j] - cl[k * H1 + lane + 32 * j];
      d += t * t;
    }
    #pragma unroll
    for (int off = 16; off > 0; off >>= 1)
      d += __shfl_xor(d, off, 32);
    float qk = 1.0f / (1.0f + d);
    qsum += qk;
    if (lane == k) qmine = qk;
  }
  if (lane < NCLUST)
    q[(size_t)row * NCLUST + lane] = qmine / qsum;
}

// ---------------------------------------------------------------------------
// Launcher
// ---------------------------------------------------------------------------
extern "C" void kernel_launch(void* const* d_in, const int* in_sizes, int n_in,
                              void* d_out, int out_size, void* d_ws, size_t ws_size,
                              hipStream_t stream) {
  (void)in_sizes; (void)n_in; (void)out_size; (void)ws_size;
  const float* x        = (const float*)d_in[0];
  const float* W1       = (const float*)d_in[1];
  const float* b1       = (const float*)d_in[2];
  const float* W2       = (const float*)d_in[3];
  const float* b2       = (const float*)d_in[4];
  const float* clusters = (const float*)d_in[5];

  float* z = (float*)d_out;                       // [N, H1]
  float* q = z + (size_t)N_ROWS * H1;             // [N, NCLUST]

  char* ws = (char*)d_ws;
  __bf16* hbuf = (__bf16*)ws;                                   // N*H0*2 = 64 MB
  __bf16* W1T  = (__bf16*)(ws + (size_t)N_ROWS * H0 * 2);       // [H0][KP1] bf16
  __bf16* W2T  = (__bf16*)(ws + (size_t)N_ROWS * H0 * 2
                              + (size_t)H0 * KP1 * 2);          // [H1][KP2] bf16

  // pre-pass: transpose + convert weights (zero-pads K..KP1)
  transpose_cvt_kernel<<<dim3(KP1 / 32, H0 / 32), 256, 0, stream>>>(W1, W1T, D_IN, H0, KP1);
  transpose_cvt_kernel<<<dim3(KP2 / 32, H1 / 32), 256, 0, stream>>>(W2, W2T, H0, H1, KP2);

  dim3 g1(H0 / BN, N_ROWS / BM);                  // (8, 256): columns fastest
  gemm1_silu_kernel<<<g1, 256, 0, stream>>>(x, W1T, b1, hbuf);

  dim3 g2(H1 / BN, N_ROWS / BM);                  // (2, 256)
  gemm2_kernel<<<g2, 256, 0, stream>>>(hbuf, W2T, b2, z);

  soft_assign_kernel<<<N_ROWS / 8, 256, 0, stream>>>(z, clusters, q);
}
